// GatedDeltaMemory_29231547416832
// MI455X (gfx1250) — matrix operand order repaired and checked by reading, not compiled, not run on hardware
//
#include <hip/hip_runtime.h>

// ---------------------------------------------------------------------------
// GatedDeltaMemory for MI455X (gfx1250, wave32, WMMA f32_16x16x32_f16 + TDM)
// B=1, T=2048, D=1024, H=16, DM=64, E=H*DM=1024
// ---------------------------------------------------------------------------

typedef __attribute__((ext_vector_type(16))) _Float16 v16h;
typedef __attribute__((ext_vector_type(8)))  _Float16 v8h;
typedef __attribute__((ext_vector_type(4)))  _Float16 v4h;
typedef __attribute__((ext_vector_type(8)))  float    v8f;
typedef __attribute__((ext_vector_type(4)))  float    v4f;
typedef __attribute__((ext_vector_type(4)))  unsigned int v4u;
typedef __attribute__((ext_vector_type(8)))  int      v8i;
typedef __attribute__((ext_vector_type(4)))  int      v4i;

static constexpr int T  = 2048;
static constexpr int D  = 1024;
static constexpr int H  = 16;
static constexpr int DM = 64;
static constexpr int E  = H * DM;   // 1024

static __device__ inline v16h cat16(v8h lo, v8h hi) {
  return __builtin_shufflevector(lo, hi, 0,1,2,3,4,5,6,7,8,9,10,11,12,13,14,15);
}

static __device__ inline v8f wmma_f16(v16h a, v16h b, v8f c) {
  return __builtin_amdgcn_wmma_f32_16x16x32_f16(
      /*neg_a=*/false, a, /*neg_b=*/false, b,
      /*c_mod=*/(short)0, c, /*reuse_a=*/false, /*reuse_b=*/false);
}

#if __has_builtin(__builtin_amdgcn_tensor_load_to_lds) && \
    __has_builtin(__builtin_amdgcn_s_wait_tensorcnt)
#define GDM_TDM 1
#endif

#ifdef GDM_TDM
// TDM load of a 2D row-major tile of f16 (tile_cols elems x tile_rows rows,
// row stride = row_stride elems) from global into contiguous LDS.
// D# bitfields per CDNA5 ISA.
// This toolchain (clang-23 / therock-10.0) exposes the 6-arg builtin:
// (uint32x4 g0, int32x8 g1, int32x4 g2, int32x4 g3, int32x8 g4, i32 cpol).
static __device__ inline void tdm_load_2d_f16(const _Float16* gsrc, unsigned lds_addr,
                                              unsigned tile_rows, unsigned tile_cols,
                                              unsigned row_stride, unsigned tensor_rows) {
  unsigned long long ga = (unsigned long long)(size_t)gsrc;
  unsigned dim0 = row_stride;      // tensor length in dim0 (elems)
  unsigned dim1 = tensor_rows;     // tensor length in dim1 (rows)
  v4u g0;
  g0[0] = 1u;                                              // count=1 (valid), no gather
  g0[1] = lds_addr;                                        // LDS byte address
  g0[2] = (unsigned)(ga & 0xFFFFFFFFu);                    // global_addr[31:0]
  g0[3] = (unsigned)((ga >> 32) & 0x01FFFFFFu) | (2u << 30); // global_addr[56:32] | type=2
  v8i g1;
  g1[0] = (int)(1u << 16);                                 // data_size=1 (2 bytes)
  g1[1] = (int)(dim0 << 16);                               // tensor_dim0[15:0]
  g1[2] = (int)((dim0 >> 16) & 0xFFFFu) | (int)(dim1 << 16);   // dim0[31:16] | dim1[15:0]
  g1[3] = (int)((dim1 >> 16) & 0xFFFFu) | (int)(tile_cols << 16); // dim1[31:16] | tile_dim0
  g1[4] = (int)tile_rows;                                  // tile_dim1 (tile_dim2=0)
  g1[5] = (int)row_stride;                                 // tensor_dim0_stride[31:0]
  g1[6] = 0;                                               // stride[47:32] | dim1_stride lo
  g1[7] = 0;
  v4i z4 = {};
  v8i z8 = {};
  __builtin_amdgcn_tensor_load_to_lds(g0, g1, z4, z4, z8, 0);
}
#endif

// ---------------------------------------------------------------------------
// fp32 -> fp16 conversion, 4 elems/thread (n must be a multiple of 4)
// ---------------------------------------------------------------------------
__global__ void gdm_cvt_f32_f16(const float* __restrict__ src,
                                _Float16* __restrict__ dst, int n) {
  int i = (blockIdx.x * blockDim.x + threadIdx.x) * 4;
  if (i < n) {
    v4f f = *(const v4f*)(src + i);
    v4h o;
#pragma unroll
    for (int e = 0; e < 4; ++e) o[e] = (_Float16)f[e];
    *(v4h*)(dst + i) = o;
  }
}

// ---------------------------------------------------------------------------
// Tiled WMMA GEMM: C[M,N] = A[M,K] * B[K,N]  (f16 in, f32 accum)
// 128 threads = 4 waves; block tile 128x64; wave tile 32x64 (8 WMMA/K-step).
// A tile fed by the Tensor Data Mover; B tile staged transposed by VALU.
// ---------------------------------------------------------------------------
template <bool OUT_F32>
__global__ __launch_bounds__(128)
void gdm_gemm_wmma(const _Float16* __restrict__ A,
                   const _Float16* __restrict__ B,
                   float* __restrict__ Cf, _Float16* __restrict__ Ch,
                   int M, int N, int K) {
  __shared__ __attribute__((aligned(32))) _Float16 As[128 * 32];
  __shared__ __attribute__((aligned(32))) _Float16 Bs[64 * 32];   // transposed [n][k]

  const int tid  = threadIdx.x;
  const int lane = tid & 31;
  const int wid  = tid >> 5;
  const int m0   = blockIdx.y * 128;
  const int n0   = blockIdx.x * 64;

  v8f acc[2][4] = {};

  const int row  = lane & 15;
  const int klo  = (lane & 16) ? 8 : 0;    // unused; A-frag also uses koff
  const int koff = (lane & 16) ? 16 : 0;   // B-frag K sub-offset (halves)

  for (int k0 = 0; k0 < K; k0 += 32) {
    __syncthreads();
#ifdef GDM_TDM
    if (wid == 0) {
      tdm_load_2d_f16(A + (size_t)m0 * K + k0, (unsigned)(size_t)&As[0],
                      /*tile_rows=*/128, /*tile_cols=*/32,
                      /*row_stride=*/(unsigned)K, /*tensor_rows=*/(unsigned)(M - m0));
    }
#else
    for (int c = tid; c < 128 * 4; c += 128) {       // 128 rows x 4 chunks of 8 halves
      int r = c >> 2, c8 = (c & 3) * 8;
      *(v8h*)&As[r * 32 + c8] = *(const v8h*)&A[(size_t)(m0 + r) * K + k0 + c8];
    }
#endif
    // Stage B tile 32x64 transposed -> Bs[n][k], vectorized global reads
    for (int c = tid; c < 32 * 8; c += 128) {        // 32 k-rows x 8 chunks of 8 cols
      int kk = c >> 3, n8 = (c & 7) * 8;
      v8h bv = *(const v8h*)&B[(size_t)(k0 + kk) * N + n0 + n8];
#pragma unroll
      for (int e = 0; e < 8; ++e) Bs[(n8 + e) * 32 + kk] = bv[e];
    }
    if (k0 + 32 < K)
      __builtin_prefetch(&B[(size_t)(k0 + 32 + (tid >> 2)) * N + n0], 0, 1);
#ifdef GDM_TDM
    if (wid == 0) __builtin_amdgcn_s_wait_tensorcnt(0);
#endif
    __syncthreads();

    // A fragments (16x32 each): lanes 0-15 rows, K {0..15}; lanes 16-31 K {16..31}
    v16h af[2];
#pragma unroll
    for (int mh = 0; mh < 2; ++mh) {
      const _Float16* ap = &As[(wid * 32 + mh * 16 + row) * 32];
      af[mh] = *(const v16h*)(ap + koff);
    }
#pragma unroll
    for (int n = 0; n < 4; ++n) {
      v16h bf = *(const v16h*)&Bs[(n * 16 + (lane & 15)) * 32 + koff];
#pragma unroll
      for (int mh = 0; mh < 2; ++mh)
        acc[mh][n] = wmma_f16(af[mh], bf, acc[mh][n]);
    }
  }

  // Write out: elem r of acc[mh][n] -> M = m0+wid*32+mh*16+r+(lane>=16?8:0), N = n0+n*16+(lane&15)
#pragma unroll
  for (int mh = 0; mh < 2; ++mh)
#pragma unroll
    for (int n = 0; n < 4; ++n)
#pragma unroll
      for (int r = 0; r < 8; ++r) {
        int mi = m0 + wid * 32 + mh * 16 + r + ((lane >> 4) << 3);
        int nj = n0 + n * 16 + (lane & 15);
        if (OUT_F32) Cf[(size_t)mi * N + nj] = acc[mh][n][r];
        else         Ch[(size_t)mi * N + nj] = (_Float16)acc[mh][n][r];
      }
}

// ---------------------------------------------------------------------------
// beta = sigmoid(x @ Wbeta), log_retain = log(1-beta+1e-8); stored [H][T]
// ---------------------------------------------------------------------------
__global__ void gdm_beta(const float* __restrict__ x,
                         const float* __restrict__ Wb,
                         float* __restrict__ beta, float* __restrict__ lr) {
  int idx = blockIdx.x * blockDim.x + threadIdx.x;  // t*H + h
  if (idx >= T * H) return;
  int t = idx / H, h = idx % H;
  float s = 0.f;
  for (int d = 0; d < D; ++d) s += x[(size_t)t * D + d] * Wb[(size_t)d * H + h];
  float b = 1.f / (1.f + __expf(-s));
  beta[h * T + t] = b;
  lr[h * T + t]   = __logf(1.f - b + 1e-8f);
}

// ---------------------------------------------------------------------------
// Per-head inclusive cumsum over T=2048 (one block per head, 8 elems/thread)
// ---------------------------------------------------------------------------
__global__ __launch_bounds__(256)
void gdm_scan(const float* __restrict__ lr, float* __restrict__ cum) {
  __shared__ float sums[256];
  int h = blockIdx.x, tid = threadIdx.x;
  float v[8], run = 0.f;
#pragma unroll
  for (int e = 0; e < 8; ++e) { v[e] = lr[h * T + tid * 8 + e]; run += v[e]; }
  sums[tid] = run;
  __syncthreads();
  for (int off = 1; off < 256; off <<= 1) {
    float add = (tid >= off) ? sums[tid - off] : 0.f;
    __syncthreads();
    sums[tid] += add;
    __syncthreads();
  }
  float acc = sums[tid] - run;  // exclusive prefix
#pragma unroll
  for (int e = 0; e < 8; ++e) { acc += v[e]; cum[h * T + tid * 8 + e] = acc; }
}

// ---------------------------------------------------------------------------
// In-place L2 norm over DM=64 per (t,h): one wave32 per vector, 2 elems/lane
// ---------------------------------------------------------------------------
__global__ __launch_bounds__(256)
void gdm_l2norm(_Float16* __restrict__ P) {
  int w = (blockIdx.x * blockDim.x + threadIdx.x) >> 5;
  int lane = threadIdx.x & 31;
  if (w >= T * H) return;
  int t = w / H, h = w % H;
  _Float16* p = P + (size_t)t * E + h * DM;
  float a = (float)p[lane * 2], b = (float)p[lane * 2 + 1];
  float ss = a * a + b * b;
#pragma unroll
  for (int m = 16; m; m >>= 1) ss += __shfl_xor(ss, m, 32);
  float sc = 1.f / fmaxf(sqrtf(ss), 1e-12f);
  p[lane * 2]     = (_Float16)(a * sc);
  p[lane * 2 + 1] = (_Float16)(b * sc);
}

// ---------------------------------------------------------------------------
// Fused decayed attention. 4 waves/block: 64 query rows of one head share a
// staged V tile. Per wave per 32-key step: 4 WMMA (scores) + 4 WMMA (W@V).
// O[i, h*64+d] = sum_{j>=i} (q_i·k_j/8)*exp(min(cum_i-cum_j,0))*v_j + beta_i*v_i
// ---------------------------------------------------------------------------
__global__ __launch_bounds__(128)
void gdm_attn(const _Float16* __restrict__ Qn, const _Float16* __restrict__ Kn,
              const _Float16* __restrict__ V,  const float* __restrict__ beta,
              const float* __restrict__ cum,   _Float16* __restrict__ O) {
  __shared__ __attribute__((aligned(32))) _Float16 Vs[64 * 32];     // V tile transposed [d][s]
  __shared__ __attribute__((aligned(32))) _Float16 Ws[4][16 * 32];  // per-wave weight tile
  __shared__ float cumI[64];
  __shared__ float betaI[64];

  const int qb = blockIdx.x, h = blockIdx.y;
  const int tid = threadIdx.x;
  const int lane = tid & 31;
  const int wid  = tid >> 5;
  const int i0   = qb * 64 + wid * 16;  // this wave's query-tile start

  if (tid < 64) {
    cumI[tid]  = cum[h * T + qb * 64 + tid];
    betaI[tid] = beta[h * T + qb * 64 + tid];
  }

  const int row  = lane & 15;
  const int klo  = (lane & 16) ? 8 : 0;
  const int koff = (lane & 16) ? 16 : 0;

  // Q A-fragments for K-dim 0..31 and 32..63 (registers for whole loop)
  const _Float16* qrow = Qn + (size_t)(i0 + row) * E + h * DM;
  v16h qa0 = *(const v16h*)(qrow + koff);
  v16h qa1 = *(const v16h*)(qrow + 32 + koff);

  v8f acc[4] = {};
  __syncthreads();

  for (int s = qb * 64; s < T; s += 32) {
    // Cooperative stage: V[s..s+31, h*64..+63] transposed -> Vs[d][ss]
    for (int c = tid; c < 32 * 8; c += 128) {   // 32 rows x 8 chunks of 8 halves
      int ss = c >> 3, d8 = (c & 7) * 8;
      v8h vv = *(const v8h*)&V[(size_t)(s + ss) * E + h * DM + d8];
#pragma unroll
      for (int e = 0; e < 8; ++e) Vs[(d8 + e) * 32 + ss] = vv[e];
    }
    __syncthreads();

    if (s + 31 >= i0) {  // wave-uniform: skip tiles entirely below the diagonal
      // Two 16x16 score sub-tiles
#pragma unroll
      for (int sub = 0; sub < 2; ++sub) {
        int jcol = s + sub * 16 + (lane & 15);
        const _Float16* krow = Kn + (size_t)jcol * E + h * DM;
        v16h kb0 = *(const v16h*)(krow + koff);        // K^T B-frag, d 0..31
        v16h kb1 = *(const v16h*)(krow + 32 + koff);   // d 32..63
        v8f qk = {};
        qk = wmma_f16(qa0, kb0, qk);
        qk = wmma_f16(qa1, kb1, qk);

        float cumJ = cum[h * T + jcol];
#pragma unroll
        for (int r = 0; r < 8; ++r) {
          int rr = r + ((lane >> 4) << 3);
          int irow = i0 + rr;
          float w = 0.f;
          if (jcol >= irow) {
            float ld = cumI[wid * 16 + rr] - cumJ;
            ld = ld < 0.f ? ld : 0.f;
            w = qk[r] * 0.125f * __expf(ld);
          }
          Ws[wid][rr * 32 + sub * 16 + (lane & 15)] = (_Float16)w;
        }
      }
      // Re-marshal weight tile as A-fragment (wave-local LDS; in-order DS)
      const _Float16* wp = &Ws[wid][row * 32];
      v16h wa = *(const v16h*)(wp + koff);
#pragma unroll
      for (int n = 0; n < 4; ++n) {
        v16h vb = *(const v16h*)&Vs[(n * 16 + (lane & 15)) * 32 + koff];
        acc[n] = wmma_f16(wa, vb, acc[n]);
      }
    }
    __syncthreads();
  }

  // Epilogue: + beta_i * v_i, write f16 O
#pragma unroll
  for (int n = 0; n < 4; ++n)
#pragma unroll
    for (int r = 0; r < 8; ++r) {
      int rr = r + ((lane >> 4) << 3);
      int irow = i0 + rr;
      int d = n * 16 + (lane & 15);
      float vv = (float)V[(size_t)irow * E + h * DM + d];
      O[(size_t)irow * E + h * DM + d] = (_Float16)(acc[n][r] + betaI[wid * 16 + rr] * vv);
    }
}

// ---------------------------------------------------------------------------
// Host launcher
// ---------------------------------------------------------------------------
extern "C" void kernel_launch(void* const* d_in, const int* in_sizes, int n_in,
                              void* d_out, int out_size, void* d_ws, size_t ws_size,
                              hipStream_t stream) {
  const float* x     = (const float*)d_in[0];
  const float* Wk    = (const float*)d_in[1];
  const float* Wv    = (const float*)d_in[2];
  const float* Wq    = (const float*)d_in[3];
  const float* Wbeta = (const float*)d_in[4];
  const float* Wout  = (const float*)d_in[5];
  float* out = (float*)d_out;

  char* ws = (char*)d_ws;
  size_t off = 0;
  auto carve = [&](size_t bytes) { char* p = ws + off; off += (bytes + 255) & ~(size_t)255; return p; };
  _Float16* xh    = (_Float16*)carve((size_t)T * D * 2);
  _Float16* Wkh   = (_Float16*)carve((size_t)D * E * 2);
  _Float16* Wvh   = (_Float16*)carve((size_t)D * E * 2);
  _Float16* Wqh   = (_Float16*)carve((size_t)D * E * 2);
  _Float16* Wouth = (_Float16*)carve((size_t)E * D * 2);
  _Float16* Kp    = (_Float16*)carve((size_t)T * E * 2);
  _Float16* Vp    = (_Float16*)carve((size_t)T * E * 2);
  _Float16* Qp    = (_Float16*)carve((size_t)T * E * 2);
  _Float16* Obuf  = (_Float16*)carve((size_t)T * E * 2);
  float* betaA = (float*)carve((size_t)H * T * 4);
  float* lrA   = (float*)carve((size_t)H * T * 4);
  float* cumA  = (float*)carve((size_t)H * T * 4);
  (void)ws_size; (void)n_in; (void)in_sizes; (void)out_size;

  // 1) fp16 conversions (4 elems/thread)
  {
    int n1 = T * D, n2 = D * E;
    gdm_cvt_f32_f16<<<(n1 / 4 + 255) / 256, 256, 0, stream>>>(x, xh, n1);
    gdm_cvt_f32_f16<<<(n2 / 4 + 255) / 256, 256, 0, stream>>>(Wk, Wkh, n2);
    gdm_cvt_f32_f16<<<(n2 / 4 + 255) / 256, 256, 0, stream>>>(Wv, Wvh, n2);
    gdm_cvt_f32_f16<<<(n2 / 4 + 255) / 256, 256, 0, stream>>>(Wq, Wqh, n2);
    gdm_cvt_f32_f16<<<(n2 / 4 + 255) / 256, 256, 0, stream>>>(Wout, Wouth, n2);
  }

  // 2) Q/K/V projections (WMMA, TDM-fed A tiles)
  {
    dim3 grid(E / 64, T / 128);
    gdm_gemm_wmma<false><<<grid, 128, 0, stream>>>(xh, Wkh, nullptr, Kp, T, E, D);
    gdm_gemm_wmma<false><<<grid, 128, 0, stream>>>(xh, Wvh, nullptr, Vp, T, E, D);
    gdm_gemm_wmma<false><<<grid, 128, 0, stream>>>(xh, Wqh, nullptr, Qp, T, E, D);
  }

  // 3) L2 normalization of K and Q
  {
    int blocks = (T * H * 32 + 255) / 256;
    gdm_l2norm<<<blocks, 256, 0, stream>>>(Kp);
    gdm_l2norm<<<blocks, 256, 0, stream>>>(Qp);
  }

  // 4) beta / log-retain, then per-head cumsum
  gdm_beta<<<(T * H + 255) / 256, 256, 0, stream>>>(x, Wbeta, betaA, lrA);
  gdm_scan<<<H, 256, 0, stream>>>(lrA, cumA);

  // 5) fused decayed attention (WMMA, 4 query tiles per block share V staging)
  {
    dim3 grid(T / 64, H);
    gdm_attn<<<grid, 128, 0, stream>>>(Qp, Kp, Vp, betaA, cumA, Obuf);
  }

  // 6) final output GEMM: out = O @ Wout (f32 result)
  {
    dim3 grid(D / 64, T / 128);
    gdm_gemm_wmma<true><<<grid, 128, 0, stream>>>(Obuf, Wouth, out, nullptr, T, D, E);
  }
}
